// CNNMnist_Sketch_7876970021370
// MI455X (gfx1250) — matrix operand order repaired and checked by reading, not compiled, not run on hardware
//
#include <hip/hip_runtime.h>
#include <hip/hip_fp16.h>

typedef __attribute__((ext_vector_type(16))) _Float16 v16h;
typedef __attribute__((ext_vector_type(8)))  float    v8f;

#define QDIM 128
#define BATCH 4096

#if __has_builtin(__builtin_amdgcn_global_load_async_to_lds_b128) && \
    __has_builtin(__builtin_amdgcn_s_wait_asynccnt)
#define USE_ASYNC_LDS 1
#define AS1 __attribute__((address_space(1)))
#define AS3 __attribute__((address_space(3)))
typedef int async_v4i __attribute__((vector_size(16)));  // matches builtin param type
#else
#define USE_ASYNC_LDS 0
#endif

// ---------------------------------------------------------------------------
// WMMA fragment loaders (CDNA5 wave32 layouts, ISA 7.12.2)
//
// A (16x32, f16), row-major source with contiguous K per row:
//   lane l (M = l&15), element e:
//     l<16 : e<8 -> K=e       ; e>=8 -> K=8+e   (16..23)
//     l>=16: e<8 -> K=8+e     ; e>=8 -> K=16+e  (24..31)
// B (32x16, f16), source row-major [N][K]:
//   lane l (N = l&15), element e: same K mapping as A
// C/D (16x16, f32): lane l (N = l&15), vgpr r: M = r + (l>=16 ? 8 : 0)
// ---------------------------------------------------------------------------
__device__ __forceinline__ v16h wmma_a_rowmajor(const _Float16* __restrict__ rowptr,
                                                int kbase, int lane) {
  const int off = kbase + ((lane & 16) ? 8 : 0);
  v16h a;
#pragma unroll
  for (int e = 0; e < 8; ++e) {
    a[e]     = rowptr[off + e];        // K = off .. off+7
    a[e + 8] = rowptr[off + 16 + e];   // K = off+16 .. off+23
  }
  return a;
}

__device__ __forceinline__ v16h wmma_b_rowmajor(const _Float16* __restrict__ w,
                                                int ldk, int n0, int kbase, int lane) {
  const int n  = n0 + (lane & 15);
  const int k0 = kbase + ((lane & 16) ? 8 : 0);
  const _Float16* p = w + (size_t)n * ldk + k0;
  v16h b;
#pragma unroll
  for (int e = 0; e < 8; ++e) { b[e] = p[e]; b[e + 8] = p[16 + e]; }
  return b;
}

// ---------------------------------------------------------------------------
// Kernel 1: reconstruct dense weights from sketches.
//   wc1f : f32 [32][25]
//   wc2h : f16 [64][800]    (OIHW flattened: ci*25 + ky*5 + kx)
//   w3h  : f16 [512][1024]
// ---------------------------------------------------------------------------
__global__ void __launch_bounds__(256) prep_weights(
    const int* __restrict__ hi1, const int* __restrict__ hi2, const int* __restrict__ hi3,
    const float* __restrict__ sg1, const float* __restrict__ sg2, const float* __restrict__ sg3,
    const float* __restrict__ w1, const float* __restrict__ w2, const float* __restrict__ w3,
    float* __restrict__ wc1f, _Float16* __restrict__ wc2h, _Float16* __restrict__ w3h) {
  const int tid = blockIdx.x * 256 + threadIdx.x;
  if (tid < 800) {                         // 32*25
    const int o = tid / 25, j = tid % 25;
    wc1f[tid] = w1[o * QDIM + hi1[j]] * sg1[j];
  } else if (tid < 800 + 51200) {          // 64*800
    const int t = tid - 800;
    const int o = t / 800, j = t % 800;
    wc2h[t] = (_Float16)(w2[o * QDIM + hi2[j]] * sg2[j]);
  } else if (tid < 800 + 51200 + 524288) { // 512*1024
    const int t = tid - 52000;
    const int o = t >> 10, j = t & 1023;
    w3h[t] = (_Float16)(w3[o * QDIM + hi3[j]] * sg3[j]);
  }
}

// ---------------------------------------------------------------------------
// Kernel 2: conv1 (1->32, 5x5, valid) + 2x2 maxpool + relu  -> h1 f16
//   h1 layout: ((b*32 + c)*12 + y)*12 + x
// ---------------------------------------------------------------------------
__global__ void __launch_bounds__(256) conv1_pool_relu(
    const float* __restrict__ x, const float* __restrict__ wc1,
    const float* __restrict__ b1, _Float16* __restrict__ h1, int total) {
  const int idx = blockIdx.x * 256 + threadIdx.x;
  if (idx >= total) return;
  int t = idx;
  const int px = t % 12; t /= 12;
  const int py = t % 12; t /= 12;
  const int c  = t % 32; t /= 32;
  const int b  = t;
  const float* __restrict__ xi = x + (size_t)b * 784;
  const float* __restrict__ w  = wc1 + c * 25;
  float m = -3.4e38f;
#pragma unroll
  for (int dy = 0; dy < 2; ++dy) {
#pragma unroll
    for (int dx = 0; dx < 2; ++dx) {
      const int oy = 2 * py + dy, ox = 2 * px + dx;
      float s = 0.0f;
#pragma unroll
      for (int ky = 0; ky < 5; ++ky)
#pragma unroll
        for (int kx = 0; kx < 5; ++kx)
          s += xi[(oy + ky) * 28 + (ox + kx)] * w[ky * 5 + kx];
      m = fmaxf(m, s);
    }
  }
  m = fmaxf(m + b1[c], 0.0f);
  h1[idx] = (_Float16)m;
}

// ---------------------------------------------------------------------------
// Kernel 3: conv2 (32->64, 5x5) as implicit-im2col WMMA GEMM, fused pool+relu.
// One block = one image; 4 waves, wave w owns M-tile w (16 of 64 positions).
// K = 800 = 25 fully-unrolled steps of 32 -> im2col LDS offsets become
// compile-time constants (selected by the lane&16 bit).
// h1 image slice staged in LDS via async global->LDS b128 when available.
// Output: flattened (B,1024) f16: b*1024 + ch*16 + py*4 + px
// ---------------------------------------------------------------------------
__global__ void __launch_bounds__(128) conv2_wmma_pool_relu(
    const _Float16* __restrict__ h1, const _Float16* __restrict__ wc2,
    const float* __restrict__ b2, _Float16* __restrict__ h2) {
  __shared__ _Float16 sH[32 * 12 * 12];   // 9216 B
  const int b   = blockIdx.x;
  const int tid = threadIdx.x;

  const _Float16* __restrict__ src = h1 + (size_t)b * 4608;
#if USE_ASYNC_LDS
  // 4608 halves = 576 x 16B chunks, ASYNCcnt-tracked DMA into LDS
#pragma unroll 1
  for (int i = tid; i < 576; i += 128) {
    __builtin_amdgcn_global_load_async_to_lds_b128(
        (AS1 async_v4i*)(src + i * 8), (AS3 async_v4i*)(sH + i * 8), 0, 0);
  }
  __builtin_amdgcn_s_wait_asynccnt(0);
  __syncthreads();
#else
  for (int i = tid; i < 4608; i += 128) sH[i] = src[i];
  __syncthreads();
#endif

  const int wave = tid >> 5;          // M-tile 0..3
  const int lane = tid & 31;
  const int m    = wave * 16 + (lane & 15);   // conv output position 0..63
  const int oy   = m >> 3, ox = m & 7;
  const int posBase = oy * 12 + ox;
  const bool hi  = (lane & 16) != 0;

  v8f acc0 = {}, acc1 = {}, acc2 = {}, acc3 = {};
#pragma unroll
  for (int kb = 0; kb < 25; ++kb) {
    const int kbb = kb * 32;
    // keep next k-block of the weight panel warm in L2/L0
    if (kb + 1 < 25) __builtin_prefetch(wc2 + (lane & 15) * 800 + kbb + 32, 0, 3);
    // A fragment: im2col gather from LDS.  With kb unrolled, the K-dependent
    // part of the LDS offset is a compile-time constant per (e, lane-half).
    v16h a;
#pragma unroll
    for (int e = 0; e < 16; ++e) {
      const int K0 = kbb + e + ((e & 8) ? 8 : 0);   // lanes 0-15
      const int K1 = K0 + 8;                        // lanes 16-31
      const int r0 = K0 % 25, r1 = K1 % 25;
      const int o0 = (K0 / 25) * 144 + (r0 / 5) * 12 + (r0 % 5);
      const int o1 = (K1 / 25) * 144 + (r1 / 5) * 12 + (r1 % 5);
      a[e] = sH[posBase + (hi ? o1 : o0)];
    }
    const v16h w0 = wmma_b_rowmajor(wc2, 800, 0,  kbb, lane);
    acc0 = __builtin_amdgcn_wmma_f32_16x16x32_f16(false, a, false, w0, (short)0, acc0, false, false);
    const v16h w1 = wmma_b_rowmajor(wc2, 800, 16, kbb, lane);
    acc1 = __builtin_amdgcn_wmma_f32_16x16x32_f16(false, a, false, w1, (short)0, acc1, false, false);
    const v16h w2 = wmma_b_rowmajor(wc2, 800, 32, kbb, lane);
    acc2 = __builtin_amdgcn_wmma_f32_16x16x32_f16(false, a, false, w2, (short)0, acc2, false, false);
    const v16h w3 = wmma_b_rowmajor(wc2, 800, 48, kbb, lane);
    acc3 = __builtin_amdgcn_wmma_f32_16x16x32_f16(false, a, false, w3, (short)0, acc3, false, false);
  }

  // Epilogue: 2x2 maxpool (in-register + cross-half shuffle) + bias + relu.
  // M-tile `wave` covers conv rows oy = 2*wave, 2*wave+1 -> pooled row py = wave.
  v8f accs[4] = {acc0, acc1, acc2, acc3};
#pragma unroll
  for (int nt = 0; nt < 4; ++nt) {
    const int ch   = nt * 16 + (lane & 15);
    const float bias = b2[ch];
#pragma unroll
    for (int px = 0; px < 4; ++px) {
      float v = fmaxf(accs[nt][2 * px], accs[nt][2 * px + 1]); // pool along x
      const float o = __shfl_xor(v, 16, 32);                    // M=r vs M=r+8 halves
      v = fmaxf(v, o);
      v = fmaxf(v + bias, 0.0f);
      if (lane < 16)
        h2[(size_t)b * 1024 + ch * 16 + wave * 4 + px] = (_Float16)v;
    }
  }
}

// ---------------------------------------------------------------------------
// Kernel 4: FC1 (4096x1024 @ 1024x512^T) WMMA GEMM + bias + relu -> h3 f32
// grid (64, 8) x 128 threads: wave owns M-tile, block.y owns 4 N-tiles.
// ---------------------------------------------------------------------------
__global__ void __launch_bounds__(128) fc1_wmma_relu(
    const _Float16* __restrict__ h2, const _Float16* __restrict__ w3h,
    const float* __restrict__ b3, float* __restrict__ h3) {
  const int tid   = threadIdx.x;
  const int wave  = tid >> 5;
  const int lane  = tid & 31;
  const int mtile = blockIdx.x * 4 + wave;   // 0..255
  const int ntb   = blockIdx.y * 4;          // N-tile base, 0..28
  const _Float16* __restrict__ arow = h2 + (size_t)(mtile * 16 + (lane & 15)) * 1024;

  v8f acc0 = {}, acc1 = {}, acc2 = {}, acc3 = {};
#pragma unroll 2
  for (int kb = 0; kb < 32; ++kb) {
    const int kbb = kb * 32;
    if (kb + 1 < 32) __builtin_prefetch(arow + kbb + 32, 0, 3);
    const v16h a  = wmma_a_rowmajor(arow, kbb, lane);
    const v16h w0 = wmma_b_rowmajor(w3h, 1024, (ntb + 0) * 16, kbb, lane);
    acc0 = __builtin_amdgcn_wmma_f32_16x16x32_f16(false, a, false, w0, (short)0, acc0, false, false);
    const v16h w1 = wmma_b_rowmajor(w3h, 1024, (ntb + 1) * 16, kbb, lane);
    acc1 = __builtin_amdgcn_wmma_f32_16x16x32_f16(false, a, false, w1, (short)0, acc1, false, false);
    const v16h w2 = wmma_b_rowmajor(w3h, 1024, (ntb + 2) * 16, kbb, lane);
    acc2 = __builtin_amdgcn_wmma_f32_16x16x32_f16(false, a, false, w2, (short)0, acc2, false, false);
    const v16h w3 = wmma_b_rowmajor(w3h, 1024, (ntb + 3) * 16, kbb, lane);
    acc3 = __builtin_amdgcn_wmma_f32_16x16x32_f16(false, a, false, w3, (short)0, acc3, false, false);
  }

  v8f accs[4] = {acc0, acc1, acc2, acc3};
#pragma unroll
  for (int j = 0; j < 4; ++j) {
    const int ch = (ntb + j) * 16 + (lane & 15);
    const float bias = b3[ch];
#pragma unroll
    for (int r = 0; r < 8; ++r) {
      const int row = mtile * 16 + r + ((lane & 16) ? 8 : 0);
      h3[(size_t)row * 512 + ch] = fmaxf(accs[j][r] + bias, 0.0f);
    }
  }
}

// ---------------------------------------------------------------------------
// Kernel 5: FC2 (512->10) + log_softmax. One thread per batch row.
// ---------------------------------------------------------------------------
__global__ void __launch_bounds__(256) fc2_logsoftmax(
    const float* __restrict__ h3, const float* __restrict__ fc2w,
    const float* __restrict__ fc2b, float* __restrict__ out) {
  const int row = blockIdx.x * 256 + threadIdx.x;
  if (row >= BATCH) return;
  const float* __restrict__ h = h3 + (size_t)row * 512;
  float lg[10];
#pragma unroll
  for (int j = 0; j < 10; ++j) lg[j] = fc2b[j];
  for (int k = 0; k < 512; ++k) {
    const float hv = h[k];
#pragma unroll
    for (int j = 0; j < 10; ++j) lg[j] += hv * fc2w[j * 512 + k];
  }
  float m = lg[0];
#pragma unroll
  for (int j = 1; j < 10; ++j) m = fmaxf(m, lg[j]);
  float s = 0.0f;
#pragma unroll
  for (int j = 0; j < 10; ++j) s += __expf(lg[j] - m);
  const float ls = __logf(s);
#pragma unroll
  for (int j = 0; j < 10; ++j) out[(size_t)row * 10 + j] = lg[j] - m - ls;
}

// ---------------------------------------------------------------------------
// Launch
// ---------------------------------------------------------------------------
extern "C" void kernel_launch(void* const* d_in, const int* in_sizes, int n_in,
                              void* d_out, int out_size, void* d_ws, size_t ws_size,
                              hipStream_t stream) {
  (void)in_sizes; (void)n_in; (void)out_size; (void)ws_size;
  // setup_inputs order:
  const float* x     = (const float*)d_in[0];   // (4096,1,28,28)
  const int*   hi1   = (const int*)  d_in[1];   // (25,)
  const int*   hi2   = (const int*)  d_in[2];   // (800,)
  const int*   hi3   = (const int*)  d_in[3];   // (1024,)
  const float* sg1   = (const float*)d_in[4];
  const float* sg2   = (const float*)d_in[5];
  const float* sg3   = (const float*)d_in[6];
  const float* w1    = (const float*)d_in[7];   // (32,128)
  const float* b1    = (const float*)d_in[8];
  const float* w2    = (const float*)d_in[9];   // (64,128)
  const float* b2    = (const float*)d_in[10];
  const float* w3    = (const float*)d_in[11];  // (512,128)
  const float* b3    = (const float*)d_in[12];
  const float* fc2w  = (const float*)d_in[13];  // (10,512)
  const float* fc2b  = (const float*)d_in[14];
  float*       out   = (float*)d_out;           // (4096,10)

  // Workspace layout (all offsets 256B-aligned), ~53 MB total.
  char* ws = (char*)d_ws;
  float*    wc1f = (float*)    (ws + 0);                 //   3,200 B (pad 4096)
  _Float16* wc2h = (_Float16*) (ws + 4096);              // 102,400 B
  _Float16* w3h  = (_Float16*) (ws + 4096 + 102400);     // 1,048,576 B
  _Float16* h1   = (_Float16*) (ws + 1155072);           // 37,748,736 B (4096*32*12*12)
  _Float16* h2   = (_Float16*) (ws + 38903808);          //  8,388,608 B (4096*1024)
  float*    h3   = (float*)    (ws + 47292416);          //  8,388,608 B (4096*512)

  // 1) weight reconstruction (800 + 51200 + 524288 = 576288 elements)
  prep_weights<<<(576288 + 255) / 256, 256, 0, stream>>>(
      hi1, hi2, hi3, sg1, sg2, sg3, w1, w2, w3, wc1f, wc2h, w3h);

  // 2) conv1 + pool + relu : 4096*32*12*12 = 18,874,368 outputs
  const int n1 = BATCH * 32 * 12 * 12;
  conv1_pool_relu<<<n1 / 256, 256, 0, stream>>>(x, wc1f, b1, h1, n1);

  // 3) conv2 WMMA GEMM + pool + relu : one block per image
  conv2_wmma_pool_relu<<<BATCH, 128, 0, stream>>>(h1, wc2h, b2, h2);

  // 4) FC1 WMMA GEMM + relu
  fc1_wmma_relu<<<dim3(64, 8), 128, 0, stream>>>(h2, w3h, b3, h3);

  // 5) FC2 + log_softmax
  fc2_logsoftmax<<<BATCH / 256, 256, 0, stream>>>(h3, fc2w, fc2b, out);
}
